// SVDNet_10960756540140
// MI455X (gfx1250) — matrix operand order repaired and checked by reading, not compiled, not run on hardware
//
#include <hip/hip_runtime.h>
#include <hip/hip_bf16.h>
#include <math.h>

typedef __attribute__((ext_vector_type(16))) _Float16 v16h;
typedef __attribute__((ext_vector_type(8)))  _Float16 v8h;
typedef __attribute__((ext_vector_type(8)))  float    v8f;

#define MDIM 512
#define NDIM 512
#define RDIM 128

// ---------------------------------------------------------------------------
// Prepack conv2/conv3 weights: OIHW f32 -> O,kh,kw,I f16 (A-fragment friendly)
// ---------------------------------------------------------------------------
__global__ void prepack_kernel(const float* __restrict__ w2,
                               const float* __restrict__ w3,
                               _Float16* __restrict__ w2p,
                               _Float16* __restrict__ w3p) {
  int i = blockIdx.x * 256 + threadIdx.x;
  if (i < 64 * 5 * 5 * 32) {                       // conv2: (64,32,5,5)
    int ic = i & 31; int r = i >> 5;
    int kw = r % 5; r /= 5;
    int kh = r % 5; int o = r / 5;
    w2p[i] = (_Float16)w2[((o * 32 + ic) * 5 + kh) * 5 + kw];
  }
  int i3 = i - 64 * 5 * 5 * 32;
  if (i3 >= 0 && i3 < 128 * 3 * 3 * 64) {          // conv3: (128,64,3,3)
    int ic = i3 & 63; int r = i3 >> 6;
    int kw = r % 3; r /= 3;
    int kh = r % 3; int o = r / 3;
    w3p[i3] = (_Float16)w3[((o * 64 + ic) * 3 + kh) * 3 + kw];
  }
}

// ---------------------------------------------------------------------------
// conv1 (2->32, 5x5, SAME) + ReLU + 2x2 maxpool, fused. Output HWC f16
// (256,256,32). Small FLOP count -> plain VALU f32.
// ---------------------------------------------------------------------------
__global__ void __launch_bounds__(256) conv1_pool_kernel(
    const float* __restrict__ x, const float* __restrict__ w,
    const float* __restrict__ b, _Float16* __restrict__ I1p) {
  __shared__ float wl[32 * 2 * 25];
  __shared__ float bl[32];
  const int t = threadIdx.x;
  for (int i = t; i < 1600; i += 256) wl[i] = w[i];
  if (t < 32) bl[t] = b[t];
  __syncthreads();

  const int c = t & 31;
  const int pix = blockIdx.x * 8 + (t >> 5);
  const int py = pix >> 8, px = pix & 255;
  float m = 0.f;  // post-ReLU values are >= 0
#pragma unroll
  for (int dy = 0; dy < 2; ++dy) {
#pragma unroll
    for (int dx = 0; dx < 2; ++dx) {
      const int oy = 2 * py + dy, ox = 2 * px + dx;
      float acc = bl[c];
#pragma unroll
      for (int ic = 0; ic < 2; ++ic) {
        const float* wc = &wl[(c * 2 + ic) * 25];
        for (int ky = 0; ky < 5; ++ky) {
          const int iy = oy + ky - 2;
          if ((unsigned)iy >= 512u) continue;
          for (int kx = 0; kx < 5; ++kx) {
            const int ix = ox + kx - 2;
            if ((unsigned)ix >= 512u) continue;
            acc += x[(iy * 512 + ix) * 2 + ic] * wc[ky * 5 + kx];
          }
        }
      }
      m = fmaxf(m, fmaxf(acc, 0.f));
    }
  }
  I1p[(py * 256 + px) * 32 + c] = (_Float16)m;
}

// ---------------------------------------------------------------------------
// conv2 (32->64, 5x5, SAME) + bias + ReLU via WMMA implicit GEMM.
// Block = 8 waves; wave tile = 16 och x 16 px; block = 64 och x 32 px in row py.
// K=32 (=Cin) per WMMA -> 25 v_wmma per wave (kh*kw), fully unrolled.
// ---------------------------------------------------------------------------
__global__ void __launch_bounds__(256) conv2_wmma_kernel(
    const _Float16* __restrict__ I1p, const _Float16* __restrict__ w2p,
    const float* __restrict__ b2, _Float16* __restrict__ O2) {
  __shared__ alignas(64) _Float16 patch[5 * 36 * 32];  // [ky][x][ic] f16
  const int t = threadIdx.x;
  const int py = blockIdx.x >> 3;
  const int px0 = (blockIdx.x & 7) << 5;

  if (t < 180) {  // 5 rows x 36 x-positions, 32 ch (64B) each
    const int r = t / 36, xo = t - r * 36;
    const int iy = py - 2 + r, ix = px0 - 2 + xo;
    uint4* dst = (uint4*)&patch[(r * 36 + xo) * 32];
    if ((unsigned)iy < 256u && (unsigned)ix < 256u) {
      const uint4* src = (const uint4*)&I1p[(iy * 256 + ix) * 32];
      dst[0] = src[0]; dst[1] = src[1]; dst[2] = src[2]; dst[3] = src[3];
    } else {
      const uint4 z = {0u, 0u, 0u, 0u};
      dst[0] = z; dst[1] = z; dst[2] = z; dst[3] = z;
    }
  }
  __syncthreads();

  const int lane = t & 31, wave = t >> 5;
  const int ochBase = (wave & 3) << 4;   // 0,16,32,48
  const int pxHalf = wave >> 2;          // 0,1
  const int mn = lane & 15, g = lane >> 4;
  const int lo = g << 3, klo = g << 4;
  const int och = ochBase + mn;
  const int pxl = (pxHalf << 4) + mn;

  v8f acc = {0.f, 0.f, 0.f, 0.f, 0.f, 0.f, 0.f, 0.f};
#pragma unroll
  for (int kh = 0; kh < 5; ++kh) {
#pragma unroll
    for (int kw = 0; kw < 5; ++kw) {
      const _Float16* wp = w2p + ((och * 5 + kh) * 5 + kw) * 32;
      v8h alo = *(const v8h*)(wp + klo);
      v8h ahi = *(const v8h*)(wp + klo + 8);
      v16h a;
#pragma unroll
      for (int e = 0; e < 8; ++e) { a[e] = alo[e]; a[e + 8] = ahi[e]; }
      v16h bfr = *(const v16h*)&patch[(kh * 36 + pxl + kw) * 32 + klo];
      acc = __builtin_amdgcn_wmma_f32_16x16x32_f16(false, a, false, bfr,
                                                   (short)0, acc, false, false);
    }
  }

  const int px = px0 + (pxHalf << 4) + mn;
  const int chb = ochBase + (g << 3);  // D rows = 8 consecutive channels
  v8h o;
#pragma unroll
  for (int r = 0; r < 8; ++r)
    o[r] = (_Float16)fmaxf(acc[r] + b2[chb + r], 0.f);
  *(v8h*)&O2[(py * 256 + px) * 64 + chb] = o;
}

// ---------------------------------------------------------------------------
// 2x2 maxpool on O2 (HWC 256x256x64) -> I3 (HWC 128x128x64)
// ---------------------------------------------------------------------------
__global__ void pool2_kernel(const _Float16* __restrict__ O2,
                             _Float16* __restrict__ I3) {
  const int idx = blockIdx.x * 256 + threadIdx.x;
  if (idx >= 128 * 128 * 64) return;
  const int c = idx & 63, p = idx >> 6;
  const int px = p & 127, py = p >> 7;
  const int base = ((2 * py) * 256 + 2 * px) * 64 + c;
  float a = (float)O2[base], b = (float)O2[base + 64];
  float cc = (float)O2[base + 256 * 64], d = (float)O2[base + 256 * 64 + 64];
  I3[idx] = (_Float16)fmaxf(fmaxf(a, b), fmaxf(cc, d));
}

// ---------------------------------------------------------------------------
// conv3 (64->128, 3x3, SAME) + bias + ReLU via WMMA. Block = 8 waves =
// 128 och x 16 px in row py. K-loop: 3x3 taps x 2 chunks of 32 ch = 18 v_wmma.
// ---------------------------------------------------------------------------
__global__ void __launch_bounds__(256) conv3_wmma_kernel(
    const _Float16* __restrict__ I3, const _Float16* __restrict__ w3p,
    const float* __restrict__ b3, _Float16* __restrict__ O3) {
  __shared__ alignas(64) _Float16 patch[3 * 18 * 64];  // [ky][x][ic]
  const int t = threadIdx.x;
  const int py = blockIdx.x >> 3;
  const int px0 = (blockIdx.x & 7) << 4;

  if (t < 54) {  // 3 rows x 18 x-positions, 64 ch (128B) each
    const int r = t / 18, xo = t - r * 18;
    const int iy = py - 1 + r, ix = px0 - 1 + xo;
    uint4* dst = (uint4*)&patch[(r * 18 + xo) * 64];
    if ((unsigned)iy < 128u && (unsigned)ix < 128u) {
      const uint4* src = (const uint4*)&I3[(iy * 128 + ix) * 64];
#pragma unroll
      for (int q = 0; q < 8; ++q) dst[q] = src[q];
    } else {
      const uint4 z = {0u, 0u, 0u, 0u};
#pragma unroll
      for (int q = 0; q < 8; ++q) dst[q] = z;
    }
  }
  __syncthreads();

  const int lane = t & 31, wave = t >> 5;
  const int ochBase = wave << 4;        // 0..112
  const int mn = lane & 15, g = lane >> 4;
  const int lo = g << 3, klo = g << 4;
  const int och = ochBase + mn;

  v8f acc = {0.f, 0.f, 0.f, 0.f, 0.f, 0.f, 0.f, 0.f};
#pragma unroll
  for (int kh = 0; kh < 3; ++kh) {
#pragma unroll
    for (int kw = 0; kw < 3; ++kw) {
#pragma unroll
      for (int kc = 0; kc < 2; ++kc) {
        const _Float16* wp = w3p + ((och * 3 + kh) * 3 + kw) * 64 + kc * 32;
        v8h alo = *(const v8h*)(wp + klo);
        v8h ahi = *(const v8h*)(wp + klo + 8);
        v16h a;
#pragma unroll
        for (int e = 0; e < 8; ++e) { a[e] = alo[e]; a[e + 8] = ahi[e]; }
        v16h bfr =
            *(const v16h*)&patch[(kh * 18 + mn + kw) * 64 + kc * 32 + klo];
        acc = __builtin_amdgcn_wmma_f32_16x16x32_f16(false, a, false, bfr,
                                                     (short)0, acc, false, false);
      }
    }
  }

  const int px = px0 + mn;
  const int chb = ochBase + (g << 3);
  v8h o;
#pragma unroll
  for (int r = 0; r < 8; ++r)
    o[r] = (_Float16)fmaxf(acc[r] + b3[chb + r], 0.f);
  *(v8h*)&O3[(py * 128 + px) * 128 + chb] = o;
}

// ---------------------------------------------------------------------------
// Global mean over 128x128 pixels per channel -> h[128]
// ---------------------------------------------------------------------------
__global__ void mean_kernel(const _Float16* __restrict__ O3,
                            float* __restrict__ h) {
  const int c = blockIdx.x, t = threadIdx.x;
  float s = 0.f;
  for (int p = t; p < 16384; p += 256) s += (float)O3[p * 128 + c];
  __shared__ float red[256];
  red[t] = s;
  __syncthreads();
  for (int off = 128; off; off >>= 1) {
    if (t < off) red[t] += red[t + off];
    __syncthreads();
  }
  if (t == 0) h[c] = red[0] * (1.0f / 16384.0f);
}

// ---------------------------------------------------------------------------
// fc: feat = relu(fc_w @ h + fc_b), 256 outputs, 1 block
// ---------------------------------------------------------------------------
__global__ void fc_kernel(const float* __restrict__ fc_w,
                          const float* __restrict__ fc_b,
                          const float* __restrict__ h,
                          float* __restrict__ feat) {
  __shared__ float hl[128];
  const int t = threadIdx.x;
  if (t < 128) hl[t] = h[t];
  __syncthreads();
  float s = fc_b[t];
  const float* wr = fc_w + t * 128;
  for (int k = 0; k < 128; ++k) s += wr[k] * hl[k];
  feat[t] = fmaxf(s, 0.f);
}

// ---------------------------------------------------------------------------
// s = softplus(s_w @ feat + s_b); bitonic sort descending; 1 block of 128
// ---------------------------------------------------------------------------
__global__ void ssort_kernel(const float* __restrict__ s_w,
                             const float* __restrict__ s_b,
                             const float* __restrict__ feat,
                             float* __restrict__ outS) {
  __shared__ float fl[256];
  __shared__ float vals[128];
  const int t = threadIdx.x;
  fl[t] = feat[t];
  fl[t + 128] = feat[t + 128];
  __syncthreads();
  float s = s_b[t];
  const float* wr = s_w + t * 256;
  for (int k = 0; k < 256; ++k) s += wr[k] * fl[k];
  vals[t] = (s > 20.f) ? s : log1pf(expf(s));
  __syncthreads();
  for (int k = 2; k <= 128; k <<= 1) {
    for (int j = k >> 1; j > 0; j >>= 1) {
      const int ixj = t ^ j;
      if (ixj > t) {
        const float a = vals[t], b = vals[ixj];
        const bool up = ((t & k) == 0);
        if ((a < b) == up) { vals[t] = b; vals[ixj] = a; }  // descending
      }
      __syncthreads();
    }
  }
  outS[t] = vals[t];
}

// ---------------------------------------------------------------------------
// u/v mat-vec: 268 MB of weights streamed, one wave per output row (HBM-bound
// part of the pipeline). float4 loads + wave shuffle reduce + prefetch.
// ---------------------------------------------------------------------------
__global__ void __launch_bounds__(256) uv_matvec_kernel(
    const float* __restrict__ u_w, const float* __restrict__ u_b,
    const float* __restrict__ v_w, const float* __restrict__ v_b,
    const float* __restrict__ feat, float* __restrict__ uraw,
    float* __restrict__ vraw) {
  __shared__ float fl[256];
  const int t = threadIdx.x;
  fl[t] = feat[t];
  __syncthreads();
  const int lane = t & 31, wave = t >> 5;
  const int row = blockIdx.x * 8 + wave;
  const float* W;
  const float* B;
  float* out;
  int r = row;
  if (row < MDIM * RDIM * 2) { W = u_w; B = u_b; out = uraw; }
  else { W = v_w; B = v_b; out = vraw; r = row - MDIM * RDIM * 2; }

  const float4* wr = (const float4*)(W + (size_t)r * 256);
  __builtin_prefetch(W + ((size_t)r + 8) * 256 + lane * 8, 0, 1);
  const float4 a0 = wr[lane * 2];
  const float4 a1 = wr[lane * 2 + 1];
  const int k = lane * 8;
  float s = a0.x * fl[k] + a0.y * fl[k + 1] + a0.z * fl[k + 2] +
            a0.w * fl[k + 3] + a1.x * fl[k + 4] + a1.y * fl[k + 5] +
            a1.z * fl[k + 6] + a1.w * fl[k + 7];
  for (int off = 16; off; off >>= 1) s += __shfl_xor(s, off, 32);
  if (lane == 0) out[r] = s + B[r];
}

// ---------------------------------------------------------------------------
// Complex Gram-Schmidt, one 512-thread block per matrix (u: 512x128, v alike).
// Per column j: parallel-CGS (equivalent to MGS for orthonormal prior q's):
//   proj_i = q_i^H v0 for all i<j (waves split i's, lanes split rows),
//   v = v0 - sum proj_i q_i, q_j = v / sqrt(|v|^2 + 1e-8).
// ---------------------------------------------------------------------------
__global__ void __launch_bounds__(512) mgs_kernel(
    const float* __restrict__ uraw, const float* __restrict__ vraw,
    float* __restrict__ outU, float* __restrict__ outV) {
  const float* raw = blockIdx.x ? vraw : uraw;
  float* Q = blockIdx.x ? outV : outU;
  __shared__ float2 v[512];
  __shared__ float2 proj[128];
  __shared__ float red[16];
  __shared__ float normsh;
  const int t = threadIdx.x, lane = t & 31, wave = t >> 5;

  for (int j = 0; j < 128; ++j) {
    v[t] = make_float2(raw[(t * 128 + j) * 2], raw[(t * 128 + j) * 2 + 1]);
    __syncthreads();

    for (int i = wave; i < j; i += 16) {
      float pr = 0.f, pi = 0.f;
      for (int r = lane; r < 512; r += 32) {
        const float qr = Q[(r * 128 + i) * 2];
        const float qi = Q[(r * 128 + i) * 2 + 1];
        const float2 vv = v[r];
        pr += qr * vv.x + qi * vv.y;  // conj(q) * v
        pi += qr * vv.y - qi * vv.x;
      }
      for (int off = 16; off; off >>= 1) {
        pr += __shfl_xor(pr, off, 32);
        pi += __shfl_xor(pi, off, 32);
      }
      if (lane == 0) { proj[i].x = pr; proj[i].y = pi; }
    }
    __syncthreads();

    float2 acc = v[t];
    for (int i = 0; i < j; ++i) {
      const float qr = Q[(t * 128 + i) * 2];
      const float qi = Q[(t * 128 + i) * 2 + 1];
      const float2 pp = proj[i];
      acc.x -= pp.x * qr - pp.y * qi;
      acc.y -= pp.x * qi + pp.y * qr;
    }
    float n2 = acc.x * acc.x + acc.y * acc.y;
    for (int off = 16; off; off >>= 1) n2 += __shfl_xor(n2, off, 32);
    if (lane == 0) red[wave] = n2;
    __syncthreads();
    if (t == 0) {
      float ss = 0.f;
      for (int w = 0; w < 16; ++w) ss += red[w];
      normsh = ss;
    }
    __syncthreads();
    const float rn = 1.0f / sqrtf(normsh + 1e-8f);
    Q[(t * 128 + j) * 2] = acc.x * rn;
    Q[(t * 128 + j) * 2 + 1] = acc.y * rn;
    __threadfence();   // make q_j visible to all waves before next column
    __syncthreads();
  }
}

// ---------------------------------------------------------------------------
extern "C" void kernel_launch(void* const* d_in, const int* in_sizes, int n_in,
                              void* d_out, int out_size, void* d_ws,
                              size_t ws_size, hipStream_t stream) {
  (void)in_sizes; (void)n_in; (void)out_size; (void)ws_size;

  const float* x       = (const float*)d_in[0];
  const float* conv1_w = (const float*)d_in[1];
  const float* conv1_b = (const float*)d_in[2];
  const float* conv2_w = (const float*)d_in[3];
  const float* conv2_b = (const float*)d_in[4];
  const float* conv3_w = (const float*)d_in[5];
  const float* conv3_b = (const float*)d_in[6];
  const float* fc_w    = (const float*)d_in[7];
  const float* fc_b    = (const float*)d_in[8];
  const float* s_w     = (const float*)d_in[9];
  const float* s_b     = (const float*)d_in[10];
  const float* u_w     = (const float*)d_in[11];
  const float* u_b     = (const float*)d_in[12];
  const float* v_w     = (const float*)d_in[13];
  const float* v_b     = (const float*)d_in[14];

  char* ws = (char*)d_ws;
  _Float16* I1p = (_Float16*)(ws + 0);                      // 4 MB (reused)
  _Float16* O2  = (_Float16*)(ws + 4194304);                // 8 MB
  _Float16* I3  = (_Float16*)(ws + 12582912);               // 2 MB
  _Float16* O3  = (_Float16*)(ws + 14680064);               // 4 MB
  _Float16* w2p = (_Float16*)(ws + 18874368);               // 100 KB
  _Float16* w3p = (_Float16*)(ws + 18976768);               // 144 KB
  float* h    = (float*)(ws + 19124224);                    // 512 B
  float* feat = (float*)(ws + 19124736);                    // 1 KB
  // raw u/v outputs alias the (dead by then) I1p region
  float* uraw = (float*)(ws + 0);                           // 512 KB
  float* vraw = (float*)(ws + 524288);                      // 512 KB

  float* out  = (float*)d_out;
  float* outU = out;                                        // 512*128*2
  float* outS = out + MDIM * RDIM * 2;                      // 128
  float* outV = out + MDIM * RDIM * 2 + RDIM;               // 512*128*2

  prepack_kernel<<<489, 256, 0, stream>>>(conv2_w, conv3_w, w2p, w3p);
  conv1_pool_kernel<<<8192, 256, 0, stream>>>(x, conv1_w, conv1_b, I1p);
  conv2_wmma_kernel<<<2048, 256, 0, stream>>>(I1p, w2p, conv2_b, O2);
  pool2_kernel<<<4096, 256, 0, stream>>>(O2, I3);
  conv3_wmma_kernel<<<1024, 256, 0, stream>>>(I3, w3p, conv3_b, O3);
  mean_kernel<<<128, 256, 0, stream>>>(O3, h);
  fc_kernel<<<1, 256, 0, stream>>>(fc_w, fc_b, h, feat);
  ssort_kernel<<<1, 128, 0, stream>>>(s_w, s_b, feat, outS);
  uv_matvec_kernel<<<32768, 256, 0, stream>>>(u_w, u_b, v_w, v_b, feat, uraw,
                                              vraw);
  mgs_kernel<<<2, 512, 0, stream>>>(uraw, vraw, outU, outV);
}
